// RegistFormer_47837345743466
// MI455X (gfx1250) — matrix operand order repaired and checked by reading, not compiled, not run on hardware
//
#include <hip/hip_runtime.h>
#include <hip/hip_bf16.h>

// ---------------------------------------------------------------------------
// RegistFormer flow-guided local attention for MI455X (gfx1250, wave32).
//   Stage 1: qp/kp/vp = Wq/Wk/Wv @ x   -> WMMA f16->f32 GEMM (1x1 conv)
//   Stage 2: bilinear flow sampling + 25-tap attention (gather/VALU bound)
//   Stage 3: out = Wfc @ attn_out      -> same WMMA GEMM kernel
// ---------------------------------------------------------------------------

typedef __attribute__((ext_vector_type(16))) _Float16 v16h;
typedef __attribute__((ext_vector_type(8)))  float    v8f;

#define KS  5
#define K2  25
#define NH  4
#define DK  16

// ---------------------------------------------------------------------------
// 1x1 conv as GEMM: Y[o,p] = sum_c Wm[o,c] * X[c,p]  (per image; X layout
// [n][C][HW]).  One wave owns a 16-pixel tile and produces all 64 output
// channels with 8 v_wmma_f32_16x16x32_f16 (4 M-tiles x 2 K-chunks of 32).
//
// Register layouts used here (wave32; A and B share contiguous K order, not the ISA table order):
//   A (16x32 f16): lane = {half,m[3:0]}, row m; VGPR v holds K-pair
//       kb = 16*half + 2v
//   B (32x16 f16): lane = {khalf,n[3:0]}, col n; VGPR v holds K-pair
//       kb = 16*khalf + 2v
//   C/D (16x16 f32): VGPR r -> row r + 8*(lane>=16), col = lane&15
// ---------------------------------------------------------------------------
__global__ __launch_bounds__(256)
void conv1x1_wmma_kernel(const float* __restrict__ Wm,
                         const float* __restrict__ X,
                         float* __restrict__ Y,
                         int HW, int C, int NPIX)
{
    const int lane = threadIdx.x & 31;
    const int wave = threadIdx.x >> 5;
    const int tile = blockIdx.x * (blockDim.x >> 5) + wave;
    if (tile * 16 >= NPIX) return;           // uniform per wave -> EXEC all-1s

    const int col  = lane & 15;
    const int half = lane >> 4;

    // tile never straddles images (HW = 16384 is a multiple of 16)
    const int  p0   = tile * 16;
    const int  nimg = p0 / HW;
    const int  rem  = p0 - nimg * HW;
    const long base = (long)nimg * C * HW + rem;

    // ---- load B (activations) for both K-chunks, f32 -> f16 ----
    v16h b0, b1;
#pragma unroll
    for (int v = 0; v < 8; ++v) {
        const int k0 = half * 16 + 2 * v;          // chunk 0: K = 0..31
        b0[2 * v]     = (_Float16)X[base + (long)(k0)      * HW + col];
        b0[2 * v + 1] = (_Float16)X[base + (long)(k0 + 1)  * HW + col];
        const int k1 = k0 + 32;                    // chunk 1: K = 32..63
        b1[2 * v]     = (_Float16)X[base + (long)(k1)      * HW + col];
        b1[2 * v + 1] = (_Float16)X[base + (long)(k1 + 1)  * HW + col];
    }

    // ---- 4 output-channel tiles of 16 ----
#pragma unroll
    for (int mt = 0; mt < 4; ++mt) {
        const int m = mt * 16 + col;               // weight row for this lane
        v16h a0, a1;
#pragma unroll
        for (int v = 0; v < 8; ++v) {
            const int kb = half * 16 + 2 * v;
            a0[2 * v]     = (_Float16)Wm[m * C + kb];
            a0[2 * v + 1] = (_Float16)Wm[m * C + kb + 1];
            a1[2 * v]     = (_Float16)Wm[m * C + kb + 32];
            a1[2 * v + 1] = (_Float16)Wm[m * C + kb + 33];
        }
        v8f c = {};
        c = __builtin_amdgcn_wmma_f32_16x16x32_f16(false, a0, false, b0,
                                                   (short)0, c, false, false);
        c = __builtin_amdgcn_wmma_f32_16x16x32_f16(false, a1, false, b1,
                                                   (short)0, c, false, false);
#pragma unroll
        for (int r = 0; r < 8; ++r) {
            const int row = mt * 16 + r + half * 8;
            Y[base + (long)row * HW + col] = c[r];
        }
    }
}

// ---------------------------------------------------------------------------
// Flow-guided 25-tap attention.  One thread per (n, head, y, x).
// Pass 1: logits[25] = (q*scale) . bilinear(kp)   (corner-major so each
//         corner's dk-dot reuses one base address)
// softmax in registers, attn written to d_out's attn region
// Pass 2: acc[dk] += attn * bilinear(vp), written to pre-fc activation.
// ---------------------------------------------------------------------------
__global__ __launch_bounds__(256)
void flow_attn_kernel(const float* __restrict__ qp,
                      const float* __restrict__ kp,
                      const float* __restrict__ vp,
                      const float* __restrict__ flow,
                      float* __restrict__ out_pre,
                      float* __restrict__ attn_out,
                      int N, int C, int H, int W)
{
    const int gid   = blockIdx.x * blockDim.x + threadIdx.x;
    const int total = N * NH * H * W;
    if (gid >= total) return;

    const int x    = gid % W;
    const int y    = (gid / W) % H;
    const int head = (gid / (W * H)) % NH;
    const int n    = gid / (W * H * NH);
    const int HWp  = H * W;
    const int cb   = head * DK;

    const float fx = flow[((n * 2 + 0) * H + y) * W + x];
    const float fy = flow[((n * 2 + 1) * H + y) * W + x];

    const float scale = 0.25f;                     // 1/sqrt(dk), dk=16
    float q[DK];
    const long qbase = ((long)(n * C + cb) * H + y) * W + x;
#pragma unroll
    for (int d = 0; d < DK; ++d) q[d] = qp[qbase + (long)d * HWp] * scale;

    const long kvbase = (long)(n * C + cb) * HWp;

    // ---- pass 1: logits ----
    float logits[K2];
    for (int kk = 0; kk < K2; ++kk) {
        const float gx  = (float)x + (float)(kk % KS - 2) + fx;
        const float gy  = (float)y + (float)(kk / KS - 2) + fy;
        const float x0f = floorf(gx), y0f = floorf(gy);
        const float wx1 = gx - x0f,   wx0 = 1.0f - wx1;
        const float wy1 = gy - y0f,   wy0 = 1.0f - wy1;
        const int   x0  = (int)x0f,   y0  = (int)y0f;

        float l = 0.0f;
        for (int cy = 0; cy < 2; ++cy) {
            const int   yi = y0 + cy;
            const float wy = cy ? wy1 : wy0;
            if (yi < 0 || yi >= H) continue;
            for (int cx = 0; cx < 2; ++cx) {
                const int   xi = x0 + cx;
                const float wx = cx ? wx1 : wx0;
                if (xi < 0 || xi >= W) continue;
                const long off = kvbase + (long)yi * W + xi;
                float dot = 0.0f;
#pragma unroll
                for (int d = 0; d < DK; ++d)
                    dot += q[d] * kp[off + (long)d * HWp];
                l += wy * wx * dot;
            }
        }
        logits[kk] = l;
    }

    // ---- softmax ----
    float mx = logits[0];
    for (int kk = 1; kk < K2; ++kk) mx = fmaxf(mx, logits[kk]);
    float s = 0.0f;
    for (int kk = 0; kk < K2; ++kk) {
        logits[kk] = __expf(logits[kk] - mx);
        s += logits[kk];
    }
    const float inv = 1.0f / s;

    // ---- pass 2: value accumulation + attn output ----
    float acc[DK];
#pragma unroll
    for (int d = 0; d < DK; ++d) acc[d] = 0.0f;

    const long abase = (((long)(n * NH + head) * K2) * H + y) * W + x;
    for (int kk = 0; kk < K2; ++kk) {
        const float a = logits[kk] * inv;
        attn_out[abase + (long)kk * HWp] = a;

        const float gx  = (float)x + (float)(kk % KS - 2) + fx;
        const float gy  = (float)y + (float)(kk / KS - 2) + fy;
        const float x0f = floorf(gx), y0f = floorf(gy);
        const float wx1 = gx - x0f,   wx0 = 1.0f - wx1;
        const float wy1 = gy - y0f,   wy0 = 1.0f - wy1;
        const int   x0  = (int)x0f,   y0  = (int)y0f;

        for (int cy = 0; cy < 2; ++cy) {
            const int   yi = y0 + cy;
            const float wy = cy ? wy1 : wy0;
            if (yi < 0 || yi >= H) continue;
            for (int cx = 0; cx < 2; ++cx) {
                const int   xi = x0 + cx;
                const float wx = cx ? wx1 : wx0;
                if (xi < 0 || xi >= W) continue;
                const long  off = kvbase + (long)yi * W + xi;
                const float wgt = a * wy * wx;
#pragma unroll
                for (int d = 0; d < DK; ++d)
                    acc[d] += wgt * vp[off + (long)d * HWp];
            }
        }
    }
#pragma unroll
    for (int d = 0; d < DK; ++d)
        out_pre[qbase + (long)d * HWp] = acc[d];
}

// ---------------------------------------------------------------------------
// launch
// ---------------------------------------------------------------------------
extern "C" void kernel_launch(void* const* d_in, const int* in_sizes, int n_in,
                              void* d_out, int out_size, void* d_ws, size_t ws_size,
                              hipStream_t stream)
{
    const float* q    = (const float*)d_in[0];
    const float* k    = (const float*)d_in[1];
    const float* v    = (const float*)d_in[2];
    const float* flow = (const float*)d_in[3];
    const float* Wq   = (const float*)d_in[4];
    const float* Wk   = (const float*)d_in[5];
    const float* Wv   = (const float*)d_in[6];
    const float* Wfc  = (const float*)d_in[7];

    const int N = 2, C = 64, H = 128, Wd = 128;
    const int HW   = H * Wd;          // 16384
    const int NPIX = N * HW;          // 32768
    const size_t TSZ = (size_t)N * C * HW;   // 2,097,152 floats per tensor

    float* ws = (float*)d_ws;
    float* qp      = ws;
    float* kp      = qp + TSZ;
    float* vp      = kp + TSZ;
    float* out_pre = vp + TSZ;        // 4 * 8 MB = 32 MB of scratch

    float* out  = (float*)d_out;      // [n, c, h, w]
    float* attn = out + TSZ;          // [n, nh, k2, h, w]

    // GEMM grid: 16-pixel tiles, 8 waves (tiles) per 256-thread block
    const int tiles   = NPIX / 16;            // 2048
    const int cblocks = tiles / 8;            // 256

    conv1x1_wmma_kernel<<<cblocks, 256, 0, stream>>>(Wq, q, qp, HW, C, NPIX);
    conv1x1_wmma_kernel<<<cblocks, 256, 0, stream>>>(Wk, k, kp, HW, C, NPIX);
    conv1x1_wmma_kernel<<<cblocks, 256, 0, stream>>>(Wv, v, vp, HW, C, NPIX);

    const int total  = N * NH * H * Wd;       // 131072 threads
    const int ablocks = (total + 255) / 256;
    flow_attn_kernel<<<ablocks, 256, 0, stream>>>(qp, kp, vp, flow,
                                                  out_pre, attn, N, C, H, Wd);

    conv1x1_wmma_kernel<<<cblocks, 256, 0, stream>>>(Wfc, out_pre, out, HW, C, NPIX);
}
